// P0GAT_77764677861541
// MI455X (gfx1250) — matrix operand order repaired and checked by reading, not compiled, not run on hardware
//
#include <hip/hip_runtime.h>
#include <hip/hip_bf16.h>
#include <math.h>

typedef __attribute__((ext_vector_type(16))) __bf16 v16bf;
typedef __attribute__((ext_vector_type(8)))  float  v8f;

#define LEAKY_SLOPE 0.2f

__device__ __forceinline__ float4 relu4(float4 v) {
  v.x = fmaxf(v.x, 0.0f); v.y = fmaxf(v.y, 0.0f);
  v.z = fmaxf(v.z, 0.0f); v.w = fmaxf(v.w, 0.0f);
  return v;
}

// ---------------------------------------------------------------------------
// GEMM: C[M x NOUT] = (RELU?)A[M x K] * B[K x NOUT], bf16 WMMA, f32 accumulate.
// One wave32 per 16-row M tile; 8 waves / block.
// B is staged in LDS pre-swizzled into WMMA fragment order so each lane's
// 16-element bf16 operand is a single contiguous 32B chunk (2x ds_load_b128).
// ---------------------------------------------------------------------------
template <int K, int NOUT, bool RELU>
__global__ void gat_gemm_wmma(const float* __restrict__ A,
                              const float* __restrict__ B,
                              float* __restrict__ C, int M) {
  constexpr int NT = NOUT / 16;   // N tiles (4 or 2)
  constexpr int KC = K / 32;      // K chunks
  constexpr int NSH = (NOUT == 64) ? 6 : 5;  // log2(NOUT)

  __shared__ v16bf Bs[KC * NT * 32];

  // Cooperative fill, source-major (coalesced global reads):
  // element B[k][n] -> frag c=k/32, t=n/16, lane l=((k/8)%2)*16 + n%16,
  //                    elem i=k%8 + ((k/16)%2)*8
  {
    __bf16* bsr = (__bf16*)Bs;
    for (int sidx = threadIdx.x; sidx < K * NOUT; sidx += blockDim.x) {
      int n = sidx & (NOUT - 1);
      int k = sidx >> NSH;
      int c = k >> 5;
      int i = (k & 7) | (((k >> 4) & 1) << 3);
      int l = ((k >> 3) & 1) * 16 + (n & 15);
      int t = n >> 4;
      int d = ((c * NT + t) * 32 + l) * 16 + i;
      bsr[d] = (__bf16)B[sidx];
    }
  }
  __syncthreads();

  const int wave = threadIdx.x >> 5;
  const int lane = threadIdx.x & 31;
  const int tileM = blockIdx.x * 8 + wave;
  const int row0 = tileM * 16;
  if (row0 >= M) return;  // whole wave exits together -> EXEC all-1s for WMMA

  const int m = lane & 15;     // A row within tile
  const int half = lane >> 4;  // lane half selects K sub-range

  v8f acc[NT] = {};
  const float* arow = A + (size_t)(row0 + m) * K;

#pragma unroll
  for (int c = 0; c < KC; ++c) {
    const int k0 = c * 32;
    // A fragment: per ISA 16-bit A layout each lane needs two runs of 8
    // consecutive floats: [k0+half*8, +8) and [k0+16+half*8, +8).
    const float4* a0 = reinterpret_cast<const float4*>(arow + k0 + half * 8);
    const float4* a1 = reinterpret_cast<const float4*>(arow + k0 + 16 + half * 8);
    float4 p0 = a0[0], p1 = a0[1];
    float4 q0 = a1[0], q1 = a1[1];
    if (RELU) { p0 = relu4(p0); p1 = relu4(p1); q0 = relu4(q0); q1 = relu4(q1); }
    v16bf a;
    a[0]  = (__bf16)p0.x; a[1]  = (__bf16)p0.y; a[2]  = (__bf16)p0.z; a[3]  = (__bf16)p0.w;
    a[4]  = (__bf16)p1.x; a[5]  = (__bf16)p1.y; a[6]  = (__bf16)p1.z; a[7]  = (__bf16)p1.w;
    a[8]  = (__bf16)q0.x; a[9]  = (__bf16)q0.y; a[10] = (__bf16)q0.z; a[11] = (__bf16)q0.w;
    a[12] = (__bf16)q1.x; a[13] = (__bf16)q1.y; a[14] = (__bf16)q1.z; a[15] = (__bf16)q1.w;

#pragma unroll
    for (int t = 0; t < NT; ++t) {
      v16bf b = Bs[(c * NT + t) * 32 + lane];  // contiguous 32B -> 2x ds_load_b128
      acc[t] = __builtin_amdgcn_wmma_f32_16x16x32_bf16(
          false, a, false, b, (short)0, acc[t], false, false);
    }
  }

  // C/D layout: VGPR r, lane l -> row = r + (l>=16)*8, col = l%16
#pragma unroll
  for (int t = 0; t < NT; ++t) {
    const int n = m + t * 16;
#pragma unroll
    for (int r = 0; r < 8; ++r) {
      int row = row0 + half * 8 + r;
      C[(size_t)row * NOUT + n] = acc[t][r];
    }
  }
}

// ---------------------------------------------------------------------------
// Per-node attention projections: as[n] = h[n].a_src ; ad[n] = h[n].a_dst
// ---------------------------------------------------------------------------
template <int F>
__global__ void node_alpha_kernel(const float* __restrict__ h,
                                  const float* __restrict__ avs,
                                  const float* __restrict__ avd,
                                  float* __restrict__ as,
                                  float* __restrict__ ad, int N) {
  int n = blockIdx.x * blockDim.x + threadIdx.x;
  if (n >= N) return;
  const float4* row = reinterpret_cast<const float4*>(h + (size_t)n * F);
  const float4* va  = reinterpret_cast<const float4*>(avs);
  const float4* vb  = reinterpret_cast<const float4*>(avd);
  float ss = 0.0f, sd = 0.0f;
#pragma unroll
  for (int i = 0; i < F / 4; ++i) {
    float4 v = row[i];
    float4 a = va[i];
    float4 b = vb[i];
    ss += v.x * a.x + v.y * a.y + v.z * a.z + v.w * a.w;
    sd += v.x * b.x + v.y * b.y + v.z * b.z + v.w * b.w;
  }
  as[n] = ss;
  ad[n] = sd;
}

// ---------------------------------------------------------------------------
// Edge scores + segment max (order-preserving uint key; key 0 == -inf/empty)
// ---------------------------------------------------------------------------
__global__ void edge_score_kernel(const int* __restrict__ ei,
                                  const float* __restrict__ as,
                                  const float* __restrict__ ad,
                                  float* __restrict__ ebuf,
                                  unsigned* __restrict__ mkey, int E) {
  int e = blockIdx.x * blockDim.x + threadIdx.x;
  if (e >= E) return;
  int s = ei[e];
  int d = ei[E + e];
  float v = as[s] + ad[d];
  v = (v > 0.0f) ? v : LEAKY_SLOPE * v;
  ebuf[e] = v;
  unsigned b = __float_as_uint(v);
  unsigned key = (b & 0x80000000u) ? ~b : (b | 0x80000000u);
  atomicMax(&mkey[d], key);
}

__global__ void node_decode_max_kernel(const unsigned* __restrict__ mkey,
                                       float* __restrict__ mf, int N) {
  int n = blockIdx.x * blockDim.x + threadIdx.x;
  if (n >= N) return;
  unsigned key = mkey[n];
  float m;
  if (key == 0u) {
    m = 0.0f;  // empty segment: reference maps -inf -> 0
  } else if (key & 0x80000000u) {
    m = __uint_as_float(key ^ 0x80000000u);
  } else {
    m = __uint_as_float(~key);
  }
  mf[n] = m;
}

__global__ void edge_exp_kernel(const int* __restrict__ ei,
                                const float* __restrict__ mf,
                                float* __restrict__ ebuf,
                                float* __restrict__ denom, int E) {
  int e = blockIdx.x * blockDim.x + threadIdx.x;
  if (e >= E) return;
  int d = ei[E + e];
  float ex = expf(ebuf[e] - mf[d]);
  ebuf[e] = ex;
  atomicAdd(&denom[d], ex);
}

__global__ void node_rdenom_kernel(float* __restrict__ denom, int N) {
  int n = blockIdx.x * blockDim.x + threadIdx.x;
  if (n >= N) return;
  float s = denom[n];
  denom[n] = (s == 0.0f) ? 1.0f : (1.0f / s);
}

// ---------------------------------------------------------------------------
// Aggregation: out[dst] += (ex/denom[dst]) * h[src]; one wave32 per edge.
// Edge index / alpha are wave-uniform -> force into SGPRs (s_load path).
// ---------------------------------------------------------------------------
template <int F>
__global__ void edge_aggregate_kernel(const int* __restrict__ ei,
                                      const float* __restrict__ ex,
                                      const float* __restrict__ rden,
                                      const float* __restrict__ h,
                                      float* __restrict__ out, int E) {
  int gid = blockIdx.x * blockDim.x + threadIdx.x;
  int lane = threadIdx.x & 31;
  int e = __builtin_amdgcn_readfirstlane(gid >> 5);  // wave-uniform edge id
  if (e >= E) return;
  int s = ei[e];
  int d = ei[E + e];
  float alpha = ex[e] * rden[d];
  const float* hs = h + (size_t)s * F;
  float* od = out + (size_t)d * F;
  atomicAdd(&od[lane], alpha * hs[lane]);
  if (F == 64) atomicAdd(&od[lane + 32], alpha * hs[lane + 32]);
}

// ---------------------------------------------------------------------------
extern "C" void kernel_launch(void* const* d_in, const int* in_sizes, int n_in,
                              void* d_out, int out_size, void* d_ws,
                              size_t ws_size, hipStream_t stream) {
  const float* x    = (const float*)d_in[0];
  const int*   ei   = (const int*)d_in[1];
  const float* W1   = (const float*)d_in[2];
  const float* a1s  = (const float*)d_in[3];
  const float* a1d  = (const float*)d_in[4];
  const float* W2   = (const float*)d_in[5];
  const float* a2s  = (const float*)d_in[6];
  const float* a2d  = (const float*)d_in[7];
  float* out = (float*)d_out;

  const int N = in_sizes[0] / 256;  // 50000
  const int E = in_sizes[1] / 2;    // 800000

  // Workspace carve-out (256B aligned slices).
  char* ws = (char*)d_ws;
  size_t off = 0;
  auto carve = [&](size_t bytes) -> void* {
    void* p = ws + off;
    off += (bytes + 255) & ~(size_t)255;
    return p;
  };
  float*    h1   = (float*)carve((size_t)N * 64 * sizeof(float));
  float*    o1   = (float*)carve((size_t)N * 64 * sizeof(float));
  float*    h2   = (float*)carve((size_t)N * 32 * sizeof(float));
  float*    as   = (float*)carve((size_t)N * sizeof(float));
  float*    ad   = (float*)carve((size_t)N * sizeof(float));
  unsigned* mkey = (unsigned*)carve((size_t)N * sizeof(unsigned));
  float*    mf   = (float*)carve((size_t)N * sizeof(float));
  float*    den  = (float*)carve((size_t)N * sizeof(float));
  float*    ebuf = (float*)carve((size_t)E * sizeof(float));
  (void)ws_size; (void)n_in; (void)out_size;

  const int tiles = (N + 15) / 16;
  dim3 gGemm((tiles + 7) / 8), bGemm(256);
  dim3 gNode((N + 255) / 256), bNode(256);
  dim3 gEdge((E + 255) / 256), bEdge(256);
  dim3 gAgg((E * 32 + 255) / 256), bAgg(256);

  // ---------------- Layer 1 ----------------
  gat_gemm_wmma<256, 64, false><<<gGemm, bGemm, 0, stream>>>(x, W1, h1, N);
  node_alpha_kernel<64><<<gNode, bNode, 0, stream>>>(h1, a1s, a1d, as, ad, N);
  hipMemsetAsync(mkey, 0, (size_t)N * sizeof(unsigned), stream);
  hipMemsetAsync(den, 0, (size_t)N * sizeof(float), stream);
  hipMemsetAsync(o1, 0, (size_t)N * 64 * sizeof(float), stream);
  edge_score_kernel<<<gEdge, bEdge, 0, stream>>>(ei, as, ad, ebuf, mkey, E);
  node_decode_max_kernel<<<gNode, bNode, 0, stream>>>(mkey, mf, N);
  edge_exp_kernel<<<gEdge, bEdge, 0, stream>>>(ei, mf, ebuf, den, E);
  node_rdenom_kernel<<<gNode, bNode, 0, stream>>>(den, N);
  edge_aggregate_kernel<64><<<gAgg, bAgg, 0, stream>>>(ei, ebuf, den, h1, o1, E);

  // ---------------- Layer 2 (ReLU fused into GEMM2 A-operand) ----------------
  gat_gemm_wmma<64, 32, true><<<gGemm, bGemm, 0, stream>>>(o1, W2, h2, N);
  node_alpha_kernel<32><<<gNode, bNode, 0, stream>>>(h2, a2s, a2d, as, ad, N);
  hipMemsetAsync(mkey, 0, (size_t)N * sizeof(unsigned), stream);
  hipMemsetAsync(den, 0, (size_t)N * sizeof(float), stream);
  hipMemsetAsync(out, 0, (size_t)N * 32 * sizeof(float), stream);
  edge_score_kernel<<<gEdge, bEdge, 0, stream>>>(ei, as, ad, ebuf, mkey, E);
  node_decode_max_kernel<<<gNode, bNode, 0, stream>>>(mkey, mf, N);
  edge_exp_kernel<<<gEdge, bEdge, 0, stream>>>(ei, mf, ebuf, den, E);
  node_rdenom_kernel<<<gNode, bNode, 0, stream>>>(den, N);
  edge_aggregate_kernel<32><<<gAgg, bAgg, 0, stream>>>(ei, ebuf, den, h2, out, E);
}
